// LogisticCoupling_14096082666044
// MI455X (gfx1250) — matrix operand order repaired and checked by reading, not compiled, not run on hardware
//
#include <hip/hip_runtime.h>
#include <hip/hip_bf16.h>

typedef __attribute__((ext_vector_type(16))) _Float16 v16h;
typedef __attribute__((ext_vector_type(8)))  _Float16 v8h;
typedef __attribute__((ext_vector_type(8)))  float    v8f;

#define HD __device__ __forceinline__

// Problem constants (from reference setup_inputs)
static constexpr int BB    = 8;
static constexpr int HH    = 64;
static constexpr int WW    = 64;
static constexpr int HWPIX = HH * WW;          // 4096
static constexpr int CH    = 48;               // ch = C/2
static constexpr int KMIX  = 8;
static constexpr int C1IN  = 144;              // ch + Cu
static constexpr int C1PAD = 160;              // K padded to multiple of 32
static constexpr int CHID  = 192;
static constexpr int C3OUT = 1248;             // ch*(2+3K)
static constexpr int C3PAD = 1280;             // Cout padded to multiple of 64
static constexpr long YOUT_ELEMS = (long)BB * 96 * HWPIX;  // 3,145,728

// ---------------------------------------------------------------------------
// Pack x_nn = concat(x[:,48:], u) -> f16 channels-last [b][pix][160], pad 0
// ---------------------------------------------------------------------------
__global__ __launch_bounds__(256) void pack_xnn_kernel(
    const float* __restrict__ x, const float* __restrict__ u,
    _Float16* __restrict__ xnn)
{
    long idx = (long)blockIdx.x * 256 + threadIdx.x;
    long total = (long)BB * HWPIX * C1PAD;
    if (idx >= total) return;
    int ci  = (int)(idx % C1PAD);
    long pl = idx / C1PAD;           // b*4096 + pix
    int pix = (int)(pl & (HWPIX - 1));
    int b   = (int)(pl >> 12);
    float v = 0.0f;
    if (ci < CH)        v = x[((long)b * 96 + (CH + ci)) * HWPIX + pix];
    else if (ci < C1IN) v = u[((long)b * 96 + (ci - CH)) * HWPIX + pix];
    xnn[idx] = (_Float16)v;
}

// ---------------------------------------------------------------------------
// Pack OIHW fp32 weights into per-fragment f16 WMMA A layout.
// Fragment (coT, tap, ciC): 512 halfs; element (lane, i):
//   m  = lane&15, hi = lane>>4
//   k  = (i<8) ? (i + hi*8) : (i + 8 + hi*8)      // ISA 16-bit A 16x32 layout
//   co = coT*16 + m ; ci = ciC*32 + k
//   value = W[co][ci][kh][kw]  (0 if ci>=Cin or co>=CoutR)
// ---------------------------------------------------------------------------
__global__ __launch_bounds__(256) void pack_w_kernel(
    const float* __restrict__ w, _Float16* __restrict__ wp,
    int Cin, int CoutR, int nCiC, long total)
{
    long idx = (long)blockIdx.x * 256 + threadIdx.x;
    if (idx >= total) return;
    int  e    = (int)(idx & 511);
    long frag = idx >> 9;
    int  ciC  = (int)(frag % nCiC);
    long tmp  = frag / nCiC;
    int  tap  = (int)(tmp % 9);
    int  coT  = (int)(tmp / 9);
    int lane = e >> 4, i = e & 15;
    int m = lane & 15, hi = lane >> 4;
    int k = (i < 8) ? (i + hi * 8) : (i + 8 + hi * 8);
    int co = coT * 16 + m;
    int ci = ciC * 32 + k;
    float v = (ci < Cin && co < CoutR) ? w[((long)co * Cin + ci) * 9 + tap] : 0.0f;
    wp[idx] = (_Float16)v;
}

// ---------------------------------------------------------------------------
// Implicit-GEMM 3x3 SAME conv via v_wmma_f32_16x16x32_f16.
// Wave tile: 32co x 32pix (2 coT x 2 pixT) -> 4 accumulators; A frags reused
// across pixel tiles, B frags reused across co tiles (1 global + 1 ds load
// per WMMA). Block: 8 waves = 2 coPair x 4 pixQuad = 64 co x 128 pixels
// (two image rows). LDS halo stage: 4 rows x 66 cols x Cpad f16.
// ---------------------------------------------------------------------------
template <bool RELU, bool F16OUT>
__global__ __launch_bounds__(256) void conv3x3_wmma_kernel(
    const _Float16* __restrict__ in,   // [b][pix][Cpad] f16
    const _Float16* __restrict__ wp,   // packed fragments (CoutP/16 coT)
    const float* __restrict__ bias,    // [CoutR]
    void* __restrict__ outp,           // f16 or f32, [b][pix][CoutP]
    int Cpad, int CoutP, int CoutR, int nCiC)
{
    extern __shared__ _Float16 smem[];   // 4*66*Cpad halfs

    const int rp  = blockIdx.x;          // row-pair index: b*32 + h0/2
    const int b   = rp >> 5;
    const int h0  = (rp & 31) << 1;
    const int tid = threadIdx.x;

    // ---- stage halo tile into LDS (coalesced 16B copies, zero pad) ----
    const int  cvecs = Cpad >> 3;                 // v8h per position
    const int  total = 4 * 66 * cvecs;
    const long inB   = (long)b * HWPIX * Cpad;
    for (int v = tid; v < total; v += 256) {
        int pos = v / cvecs, cv = v - pos * cvecs;
        int r = pos / 66, wl = pos - r * 66;
        int ih = h0 + r - 1, iw = wl - 1;
        v8h val = {};
        if ((unsigned)ih < 64u && (unsigned)iw < 64u)
            val = *(const v8h*)(in + inB + (long)(ih * 64 + iw) * Cpad + cv * 8);
        *(v8h*)(smem + (long)pos * Cpad + cv * 8) = val;
    }
    __syncthreads();

    // ---- per-wave 32co x 32pix register-blocked WMMA tile ----
    const int wave   = tid >> 5;
    const int lane   = tid & 31;
    const int coPair = wave >> 2;        // 0..1
    const int pq     = wave & 3;         // 0..3
    const int hrow   = pq >> 1;          // which of the two output rows
    const int wbase  = (pq & 1) * 32;    // 32-pixel span within the row
    const int n      = lane & 15;
    const int hi     = lane >> 4;

    const int coTbase = blockIdx.y * 4 + coPair * 2;   // 2 coT per wave

    v8f acc00 = {}, acc01 = {}, acc10 = {}, acc11 = {};

    const long fragsPerCoT = (long)9 * nCiC * 512;
    const _Float16* wpA0 = wp + (long)(coTbase + 0) * fragsPerCoT + lane * 16;
    const _Float16* wpA1 = wp + (long)(coTbase + 1) * fragsPerCoT + lane * 16;

    for (int tap = 0; tap < 9; ++tap) {
        const int kh = tap / 3;
        const int kw = tap - kh * 3;
        const long rb0 = (long)((hrow + kh) * 66 + wbase + n + kw) * Cpad;
        const long rb1 = rb0 + (long)16 * Cpad;       // +16 pixels
        const long ftap = (long)tap * nCiC * 512;
        for (int cc = 0; cc < nCiC; ++cc) {
            const long fo = ftap + (long)cc * 512;
            __builtin_prefetch((const void*)(wpA0 + fo + 512), 0, 1);
            __builtin_prefetch((const void*)(wpA1 + fo + 512), 0, 1);
            v16h a0 = *(const v16h*)(wpA0 + fo);
            v16h a1 = *(const v16h*)(wpA1 + fo);

            const int cof = cc * 32 + hi * 8;
            const _Float16* bp0 = smem + rb0 + cof;
            const _Float16* bp1 = smem + rb1 + cof;
            v8h b0lo = *(const v8h*)bp0;
            v8h b0hi = *(const v8h*)(bp0 + 16);
            v8h b1lo = *(const v8h*)bp1;
            v8h b1hi = *(const v8h*)(bp1 + 16);
            v16h b0, b1;
#pragma unroll
            for (int i = 0; i < 8; ++i) {
                b0[i] = b0lo[i]; b0[i + 8] = b0hi[i];
                b1[i] = b1lo[i]; b1[i + 8] = b1hi[i];
            }
            acc00 = __builtin_amdgcn_wmma_f32_16x16x32_f16(false, a0, false, b0, (short)0, acc00, false, false);
            acc01 = __builtin_amdgcn_wmma_f32_16x16x32_f16(false, a0, false, b1, (short)0, acc01, false, false);
            acc10 = __builtin_amdgcn_wmma_f32_16x16x32_f16(false, a1, false, b0, (short)0, acc10, false, false);
            acc11 = __builtin_amdgcn_wmma_f32_16x16x32_f16(false, a1, false, b1, (short)0, acc11, false, false);
        }
    }

    // ---- bias (+relu), channels-last stores: contiguous 8 outputs/lane ----
    const v8f* accs[4] = { &acc00, &acc01, &acc10, &acc11 };
#pragma unroll
    for (int ct = 0; ct < 2; ++ct) {
#pragma unroll
        for (int t = 0; t < 2; ++t) {
            const v8f& a = *accs[ct * 2 + t];
            const int  co0  = (coTbase + ct) * 16 + hi * 8;
            const long pixG = (long)b * HWPIX + (h0 + hrow) * 64 + wbase + t * 16 + n;
            if constexpr (F16OUT) {
                v8h o;
#pragma unroll
                for (int j = 0; j < 8; ++j) {
                    float bv = (co0 + j < CoutR) ? bias[co0 + j] : 0.0f;
                    float tv = a[j] + bv;
                    if (RELU) tv = fmaxf(tv, 0.0f);
                    o[j] = (_Float16)tv;
                }
                *(v8h*)((_Float16*)outp + pixG * CoutP + co0) = o;
            } else {
                float* op = (float*)outp + pixG * CoutP + co0;
#pragma unroll
                for (int j = 0; j < 8; ++j) {
                    float bv = (co0 + j < CoutR) ? bias[co0 + j] : 0.0f;
                    float tv = a[j] + bv;
                    if (RELU) tv = fmaxf(tv, 0.0f);
                    op[j] = tv;
                }
            }
        }
    }
}

// ---------------------------------------------------------------------------
// Logistic-mixture epilogue + per-batch ldj reduction
// ---------------------------------------------------------------------------
HD float softplusf_(float x) {
    return (x > 0.0f) ? (x + log1pf(expf(-x))) : log1pf(expf(x));
}
HD float logsigf_(float x) { return -softplusf_(-x); }

__global__ __launch_bounds__(256) void epilogue_kernel(
    const float* __restrict__ x,      // [8][96][4096] fp32 NCHW
    const float* __restrict__ nn,     // [8*4096][1280] fp32 channels-last (padded)
    const float* __restrict__ sf_in,  // [48]
    float* __restrict__ yout,         // [8][96][4096]
    float* __restrict__ ldj)          // [8]
{
    __shared__ float red[256];
    const int tid = threadIdx.x;
    long idx = (long)blockIdx.x * 256 + tid;   // over 8*48*4096; block-uniform (b,c)
    const int b   = (int)(idx / ((long)CH * HWPIX));
    const int rem = (int)(idx % ((long)CH * HWPIX));
    const int c   = rem >> 12;
    const int pix = rem & 4095;

    const float xa = x[((long)b * 96 + c) * HWPIX + pix];
    const float xb = x[((long)b * 96 + CH + c) * HWPIX + pix];
    const float* nb = nn + ((long)b * HWPIX + pix) * C3PAD;

    float log_s = nb[c];
    float t     = nb[CH + c];

    float lp[KMIX], z[KMIX], sc[KMIX];
    float mx = -1e30f;
#pragma unroll
    for (int k = 0; k < KMIX; ++k) {
        lp[k] = nb[(2 + k) * CH + c];
        mx = fmaxf(mx, lp[k]);
    }
    float se = 0.0f;
#pragma unroll
    for (int k = 0; k < KMIX; ++k) se += expf(lp[k] - mx);
    const float lse = mx + logf(se);
#pragma unroll
    for (int k = 0; k < KMIX; ++k) {
        lp[k] -= lse;
        float mu = nb[(2 + KMIX + k) * CH + c];
        sc[k] = fmaxf(nb[(2 + 2 * KMIX + k) * CH + c], -7.0f);
        z[k]  = (xa - mu) * expf(-sc[k]);
    }

    const float sfac = expf(sf_in[c]);
    log_s = tanhf(log_s / sfac) * sfac;
    const float s = expf(log_s);

    // log_cdf = LSE_k(lp + log_sigmoid(z))
    float ak[KMIX], m2 = -1e30f;
#pragma unroll
    for (int k = 0; k < KMIX; ++k) { ak[k] = lp[k] + logsigf_(z[k]); m2 = fmaxf(m2, ak[k]); }
    float s2 = 0.0f;
#pragma unroll
    for (int k = 0; k < KMIX; ++k) s2 += expf(ak[k] - m2);
    const float lcdf = m2 + logf(s2);

    const float y      = expf(lcdf);
    const float lgy    = logf(fmaxf(y, 1e-22f));
    const float lg1my  = logf(fmaxf(1.0f - y, 1e-22f));
    const float ylogit = lgy - lg1my;
    const float ldjs   = -lgy - lg1my;
    const float ya     = (ylogit + t) * s;

    // ldj_log = LSE_k(lp + z - sc - 2*softplus(z))
    float ck[KMIX], m3 = -1e30f;
#pragma unroll
    for (int k = 0; k < KMIX; ++k) {
        ck[k] = lp[k] + z[k] - sc[k] - 2.0f * softplusf_(z[k]);
        m3 = fmaxf(m3, ck[k]);
    }
    float s3 = 0.0f;
#pragma unroll
    for (int k = 0; k < KMIX; ++k) s3 += expf(ck[k] - m3);
    const float ldjl = m3 + logf(s3);

    yout[((long)b * 96 + c) * HWPIX + pix]      = ya;
    yout[((long)b * 96 + CH + c) * HWPIX + pix] = xb;

    red[tid] = ldjs + ldjl + s;
    __syncthreads();
#pragma unroll
    for (int off = 128; off > 0; off >>= 1) {
        if (tid < off) red[tid] += red[tid + off];
        __syncthreads();
    }
    if (tid == 0) atomicAdd(&ldj[b], red[0]);
}

// ---------------------------------------------------------------------------
extern "C" void kernel_launch(void* const* d_in, const int* in_sizes, int n_in,
                              void* d_out, int out_size, void* d_ws, size_t ws_size,
                              hipStream_t stream) {
    (void)in_sizes; (void)n_in; (void)out_size; (void)ws_size;

    const float* x   = (const float*)d_in[0];
    const float* u   = (const float*)d_in[1];
    const float* w1  = (const float*)d_in[3];
    const float* b1  = (const float*)d_in[4];
    const float* w2  = (const float*)d_in[5];
    const float* b2  = (const float*)d_in[6];
    const float* w3  = (const float*)d_in[7];
    const float* b3  = (const float*)d_in[8];
    const float* sfc = (const float*)d_in[9];

    // workspace carve-out
    size_t off = 0;
    auto carve = [&](size_t bytes) -> char* {
        char* p = (char*)d_ws + off;
        off = (off + bytes + 255) & ~(size_t)255;
        return p;
    };
    const long nXNN = (long)BB * HWPIX * C1PAD;
    const long nH   = (long)BB * HWPIX * CHID;
    const long nNN3 = (long)BB * HWPIX * C3PAD;
    const long nWP1 = (long)(CHID / 16)  * 9 * (C1PAD / 32) * 512;   // 276,480
    const long nWP2 = (long)(CHID / 16)  * 9 * (CHID / 32) * 512;    // 331,776
    const long nWP3 = (long)(C3PAD / 16) * 9 * (CHID / 32) * 512;    // 2,211,840

    _Float16* xnn = (_Float16*)carve(nXNN * 2);
    _Float16* wp1 = (_Float16*)carve(nWP1 * 2);
    _Float16* wp2 = (_Float16*)carve(nWP2 * 2);
    _Float16* wp3 = (_Float16*)carve(nWP3 * 2);
    _Float16* h1  = (_Float16*)carve(nH * 2);
    _Float16* h2  = (_Float16*)carve(nH * 2);
    float*    nn3 = (float*)   carve(nNN3 * 4);

    // zero ldj accumulators (tail of d_out)
    hipMemsetAsync((char*)d_out + YOUT_ELEMS * sizeof(float), 0, BB * sizeof(float), stream);

    // pack activations & weights
    pack_xnn_kernel<<<dim3((unsigned)((nXNN + 255) / 256)), 256, 0, stream>>>(x, u, xnn);
    pack_w_kernel<<<dim3((unsigned)((nWP1 + 255) / 256)), 256, 0, stream>>>(w1, wp1, C1IN, CHID,  C1PAD / 32, nWP1);
    pack_w_kernel<<<dim3((unsigned)((nWP2 + 255) / 256)), 256, 0, stream>>>(w2, wp2, CHID, CHID,  CHID / 32, nWP2);
    pack_w_kernel<<<dim3((unsigned)((nWP3 + 255) / 256)), 256, 0, stream>>>(w3, wp3, CHID, C3OUT, CHID / 32, nWP3);

    // conv chain (grid.x = B*H/2 row-pairs, grid.y = CoutP/64)
    const unsigned rowPairs = BB * HH / 2;   // 256
    conv3x3_wmma_kernel<true, true><<<dim3(rowPairs, CHID / 64), 256,
        4 * 66 * C1PAD * sizeof(_Float16), stream>>>(
        xnn, wp1, b1, (void*)h1, C1PAD, CHID, CHID, C1PAD / 32);
    conv3x3_wmma_kernel<true, true><<<dim3(rowPairs, CHID / 64), 256,
        4 * 66 * CHID * sizeof(_Float16), stream>>>(
        h1, wp2, b2, (void*)h2, CHID, CHID, CHID, CHID / 32);
    conv3x3_wmma_kernel<false, false><<<dim3(rowPairs, C3PAD / 64), 256,
        4 * 66 * CHID * sizeof(_Float16), stream>>>(
        h2, wp3, b3, (void*)nn3, CHID, C3PAD, C3OUT, CHID / 32);

    // logistic mixture epilogue + ldj reduction
    const long nEpi = (long)BB * CH * HWPIX;   // 1,572,864
    epilogue_kernel<<<dim3((unsigned)(nEpi / 256)), 256, 0, stream>>>(
        x, nn3, sfc, (float*)d_out, (float*)d_out + YOUT_ELEMS);
}
